// TrajsEncoder_39152921870521
// MI455X (gfx1250) — matrix operand order repaired and checked by reading, not compiled, not run on hardware
//
#include <hip/hip_runtime.h>

// ---------------- problem constants (from reference) ----------------
#define GG    1024            // graphs
#define PP    256             // nodes per graph (contiguous per graph)
#define DEG   8
#define EPG   (PP * DEG)      // 2048 edges per graph (contiguous per graph)
#define EPSN  1e-5f

typedef __attribute__((ext_vector_type(16))) _Float16 v16h;
typedef __attribute__((ext_vector_type(8)))  float    v8f;
typedef _Float16 h16;

// weight pointer indices inside K1Args::W  (W[i] == d_in[6+i], params pytree in
// insertion order: nodes_mlp, edges_mlp, att_nn, gine1_nn, gine1_lin, gine2_nn,
// gine2_lin, gate_nn; "final" weights (d_in[44..47]) are used by the small kernels)
enum {
  iNW0 = 0, iNB0, iNW1, iNB1, iNW2, iNB2,     // nodes_mlp  [8,32][32,32][32,8]
  iEW0,     iEB0, iEW1, iEB1, iEW2, iEB2,     // edges_mlp  [8,32][32,32][32,8]
  iAW0,     iAB0, iAW1, iAB1, iAW2, iAB2,     // att_nn     [8,32][32,32][32,64]
  iG1W0, iG1B0, iG1W1, iG1B1, iG1W2, iG1B2,   // gine1_nn   [64,32][32,32][32,64]
  iL1W,  iL1B,                                // gine1_lin  [8,64]
  iG2W0, iG2B0, iG2W1, iG2B1, iG2W2, iG2B2,   // gine2_nn
  iL2W,  iL2B,                                // gine2_lin
  iGaW0, iGaB0, iGaW1, iGaB1,                 // gate_nn    [192,32][32,1]
  NW_TOTAL                                    // == 38
};

struct K1Args {
  const float* x;      // [N,8]
  const float* ea;     // [E,8]
  const int*   col;    // [E]
  const float* W[NW_TOTAL];
  float*       pooled; // [G,192] out
};

// ---------------- dynamic LDS layout (bytes) ----------------
constexpr int oAcc  = 0;                 // float [256*64]  (doubles as f32 [2048*8])
constexpr int oMsg  = oAcc  + 65536;     // h16   [256*64]  f16 A-staging / lin(e) tile
constexpr int oTmp  = oMsg  + 32768;     // h16   [256*64]  MLP hidden ping/pong halves
constexpr int oE    = oTmp  + 32768;     // h16   [2048*8]  edge embeddings
constexpr int oX1   = oE    + 32768;     // h16   [3][256*64] x1,x2,x3 contiguous
constexpr int oX0   = oX1   + 3*32768;   // h16   [256*8]
constexpr int oW    = oX0   + 4096;      // h16   [12288]  packed B fragments (zero-padded)
constexpr int oB    = oW    + 24576;     // float [256]    staged biases (+small vecs)
constexpr int oCol  = oB    + 1024;      // int   [2048]   graph-local col indices
constexpr int oRed  = oCol  + 8192;      // float [512]    reduction scratch / stats
constexpr int oGate = oRed  + 2048;      // float [256]    gate / softmax weights
constexpr int LDS_BYTES = oGate + 1024;  // 303104 B < 320 KB per WGP

constexpr int XSTRIDE = 256 * 64;        // h16 elements per x-buffer

// ---------------- WMMA fragment handling (ISA 7.12.2 layouts, wave32) ----------------
// A: 16x32 f16 tile. lane L: M=L%16, K-half offset=(L/16)*8.
// half i: K = kt*32 + (i&7) + ((i>>3)<<4) + ((L>>4)<<3)
template <int KACT, int LD>
__device__ __forceinline__ v16h loadAfrag(const h16* A, int mt, int kt, int lane) {
  v16h a;
  int m  = mt * 16 + (lane & 15);
  int kb = kt * 32 + ((lane >> 4) << 3);
  if constexpr (KACT % 32 == 0) {
#pragma unroll
    for (int i = 0; i < 16; ++i) {
      int k = kb + (i & 7) + ((i >> 3) << 4);
      a[i] = A[m * LD + k];
    }
  } else {
    static_assert(KACT == 8, "only K=8 tail supported");
#pragma unroll
    for (int i = 0; i < 16; ++i) {
      int k = kb + (i & 7) + ((i >> 3) << 4);
      h16 v = A[m * LD + (k & (KACT - 1))];   // clamped: unconditional load
      a[i] = (k < KACT) ? v : (h16)0.f;       // per-lane select, no branch
    }
  }
  return a;
}

// Pack row-major f32 W[K,N] into per-fragment f16 layout:
// dst[((nt*KT+kt)*32 + lane)*16 + i], zero-padded for k>=KACT / n>=NACT.
template <int KACT, int NACT>
__device__ __forceinline__ void packB(h16* dst, const float* src, int tid) {
  constexpr int NT = (NACT + 15) / 16;
  constexpr int KT = (KACT + 31) / 32;
  constexpr int TOT = NT * KT * 512;
  for (int idx = tid; idx < TOT; idx += 256) {
    int tile = idx >> 9;                 // nt*KT + kt
    int lane = (idx >> 4) & 31;
    int i    = idx & 15;
    int nt = tile / KT, kt = tile % KT;
    int n  = nt * 16 + (lane & 15);
    int k  = kt * 32 + ((lane >> 4) << 3) + (i & 7) + ((i >> 3) << 4);
    float v = (n < NACT && k < KACT) ? src[k * NACT + n] : 0.f;
    dst[idx] = (h16)v;
  }
}

// B fragment: one aligned 32-byte LDS vector load, no guards.
template <int KACT, int NACT>
__device__ __forceinline__ v16h loadBfrag(const h16* Bp, int nt, int kt, int lane) {
  constexpr int KT = (KACT + 31) / 32;
  return *(const v16h*)(Bp + (((nt * KT) + kt) << 9) + (lane << 4));
}

// out[256,NACT] = act(A[256,KACT] @ W + bias). 8 waves, uniform constant tile loop
// (EXEC stays all-ones at every WMMA).
template <int KACT, int NACT, bool RELU, typename AF>
__device__ __forceinline__ void dense256(AF loadA, const h16* Bp, const float* bias,
                                         h16* outH, float* outF, int ldOut, int tid) {
  constexpr int NT = (NACT + 15) / 16;
  constexpr int KT = (KACT + 31) / 32;
  constexpr int PER_WAVE = (16 * NT) / 8;
  const int lane = tid & 31, wv = tid >> 5;
#pragma unroll
  for (int it = 0; it < PER_WAVE; ++it) {
    int t = wv + it * 8;
    int mt = t & 15, nt = t >> 4;
    v8f c = {};
#pragma unroll
    for (int kt = 0; kt < KT; ++kt) {
      v16h a = loadA(mt, kt, lane);
      v16h b = loadBfrag<KACT, NACT>(Bp, nt, kt, lane);
      c = __builtin_amdgcn_wmma_f32_16x16x32_f16(false, a, false, b,
                                                 (short)0, c, false, false);
    }
    int n = nt * 16 + (lane & 15);
    if (NACT >= 16 || n < NACT) {
      float bb = bias[n];
      int mb = mt * 16 + ((lane >> 4) << 3);
#pragma unroll
      for (int r = 0; r < 8; ++r) {
        float v = c[r] + bb;
        if (RELU) v = fmaxf(v, 0.f);
        if (outH) outH[(mb + r) * ldOut + n] = (h16)v;
        else      outF[(mb + r) * ldOut + n] = v;
      }
    }
  }
}

// per-graph instance-norm stats: mean -> sRed[c], rstd -> sRed[C+c]
template <typename T>
__device__ __forceinline__ void inormStats(const T* data, int M, int C, int ld,
                                           float* sRed, int tid) {
  __syncthreads();
  int Gn = 256 / C;
  int c = tid % C, grp = tid / C;
  float s = 0.f, q = 0.f;
  for (int m = grp; m < M; m += Gn) {
    float v = (float)data[m * ld + c];
    s += v; q += v * v;
  }
  sRed[grp * 2 * C + c]     = s;
  sRed[grp * 2 * C + C + c] = q;
  __syncthreads();
  if (tid < C) {
    float ts = 0.f, tq = 0.f;
    for (int g2 = 0; g2 < Gn; ++g2) {
      ts += sRed[g2 * 2 * C + tid];
      tq += sRed[g2 * 2 * C + C + tid];
    }
    float mean = ts / (float)M;
    float var  = tq / (float)M - mean * mean;
    sRed[tid]     = mean;
    sRed[C + tid] = rsqrtf(var + EPSN);
  }
  __syncthreads();
}

__device__ __forceinline__ void copyW(float* dst, const float* src, int n, int tid) {
  for (int i = tid; i < n; i += 256) dst[i] = src[i];
}

// ================= kernel 1: whole per-graph pipeline, 1 block = 1 graph =================
__global__ __launch_bounds__(256) void trajs_graph_kernel(K1Args a) {
  extern __shared__ __align__(32) char smem[];
  float* sAcc  = (float*)(smem + oAcc);
  h16*   sMsg  = (h16*)(smem + oMsg);
  h16*   sTmp  = (h16*)(smem + oTmp);
  h16*   sE    = (h16*)(smem + oE);
  h16*   sXb   = (h16*)(smem + oX1);     // x1|x2|x3, each XSTRIDE elements
  h16*   sX0   = (h16*)(smem + oX0);
  h16*   sWp   = (h16*)(smem + oW);      // packed B fragments
  float* sB    = (float*)(smem + oB);
  int*   sCol  = (int*)(smem + oCol);
  float* sRed  = (float*)(smem + oRed);
  float* sGate = (float*)(smem + oGate);

  const int tid = threadIdx.x;
  const int g   = blockIdx.x;

  __builtin_prefetch(a.x + (size_t)g * PP * 8, 0, 1);
  __builtin_prefetch(a.ea + (size_t)g * EPG * 8, 0, 1);

  // graph-local col indices
  for (int i = tid; i < EPG; i += 256) sCol[i] = a.col[(size_t)g * EPG + i] - g * PP;

  // ---------- edge path: inorm -> edges_mlp -> inorm -> sE (f16 [2048,8]) ----------
  for (int i = tid; i < EPG * 8; i += 256) sAcc[i] = a.ea[(size_t)g * EPG * 8 + i];
  inormStats(sAcc, EPG, 8, 8, sRed, tid);
  for (int i = tid; i < EPG * 8; i += 256) {   // normalized, f16, -> sMsg [2048,8]
    int c = i & 7;
    sMsg[i] = (h16)((sAcc[i] - sRed[c]) * sRed[8 + c]);
  }
  packB<8, 32>(sWp,        a.W[iEW0], tid);    // 1024 halves
  packB<32, 32>(sWp + 1024, a.W[iEW1], tid);   // 1024
  packB<32, 8>(sWp + 2048, a.W[iEW2], tid);    // 512
  copyW(sB,      a.W[iEB0], 32, tid);
  copyW(sB + 32, a.W[iEB1], 32, tid);
  copyW(sB + 64, a.W[iEB2], 8,  tid);
  __syncthreads();
  for (int ch = 0; ch < 8; ++ch) {             // 8 chunks of 256 edges
    const h16* Ain = sMsg + ch * 256 * 8;
    dense256<8, 32, true>([&](int mt, int kt, int ln) -> v16h {
               return loadAfrag<8, 8>(Ain, mt, kt, ln); },
             sWp, sB, sTmp, nullptr, 64, tid);
    __syncthreads();
    dense256<32, 32, true>([&](int mt, int kt, int ln) -> v16h {
               return loadAfrag<32, 64>(sTmp, mt, kt, ln); },
             sWp + 1024, sB + 32, sTmp + 32, nullptr, 64, tid);
    __syncthreads();
    dense256<32, 8, false>([&](int mt, int kt, int ln) -> v16h {
               return loadAfrag<32, 64>(sTmp + 32, mt, kt, ln); },
             sWp + 2048, sB + 64, sE + ch * 256 * 8, nullptr, 8, tid);
    __syncthreads();
  }
  inormStats(sE, EPG, 8, 8, sRed, tid);
  for (int i = tid; i < EPG * 8; i += 256) {
    int c = i & 7;
    sE[i] = (h16)(((float)sE[i] - sRed[c]) * sRed[8 + c]);
  }

  // ---------- node path: inorm -> nodes_mlp -> x0 (f16 [256,8]) ----------
  __syncthreads();
  for (int i = tid; i < PP * 8; i += 256) sAcc[i] = a.x[(size_t)g * PP * 8 + i];
  inormStats(sAcc, PP, 8, 8, sRed, tid);
  for (int i = tid; i < PP * 8; i += 256) {    // normalized f16 -> sMsg [256,8]
    int c = i & 7;
    sMsg[i] = (h16)((sAcc[i] - sRed[c]) * sRed[8 + c]);
  }
  packB<8, 32>(sWp,        a.W[iNW0], tid);
  packB<32, 32>(sWp + 1024, a.W[iNW1], tid);
  packB<32, 8>(sWp + 2048, a.W[iNW2], tid);
  copyW(sB,      a.W[iNB0], 32, tid);
  copyW(sB + 32, a.W[iNB1], 32, tid);
  copyW(sB + 64, a.W[iNB2], 8,  tid);
  __syncthreads();
  dense256<8, 32, true>([&](int mt, int kt, int ln) -> v16h {
             return loadAfrag<8, 8>(sMsg, mt, kt, ln); },
           sWp, sB, sTmp, nullptr, 64, tid);
  __syncthreads();
  dense256<32, 32, true>([&](int mt, int kt, int ln) -> v16h {
             return loadAfrag<32, 64>(sTmp, mt, kt, ln); },
           sWp + 1024, sB + 32, sTmp + 32, nullptr, 64, tid);
  __syncthreads();
  dense256<32, 8, false>([&](int mt, int kt, int ln) -> v16h {
             return loadAfrag<32, 64>(sTmp + 32, mt, kt, ln); },
           sWp + 2048, sB + 64, sX0, nullptr, 8, tid);
  __syncthreads();

  // ---------- GIN: x1 = inorm(att_nn(x0 + sum_j x0[col])) ----------
  for (int i = tid; i < PP * 8; i += 256) {    // agg+x0, f16 -> sMsg [256,8]
    int n = i >> 3, c = i & 7;
    float s = (float)sX0[i];
#pragma unroll
    for (int d = 0; d < DEG; ++d) s += (float)sX0[sCol[n * DEG + d] * 8 + c];
    sMsg[i] = (h16)s;
  }
  packB<8, 32>(sWp,        a.W[iAW0], tid);
  packB<32, 32>(sWp + 1024, a.W[iAW1], tid);
  packB<32, 64>(sWp + 2048, a.W[iAW2], tid);   // 2048 halves
  copyW(sB,      a.W[iAB0], 32, tid);
  copyW(sB + 32, a.W[iAB1], 32, tid);
  copyW(sB + 64, a.W[iAB2], 64, tid);
  __syncthreads();
  dense256<8, 32, true>([&](int mt, int kt, int ln) -> v16h {
             return loadAfrag<8, 8>(sMsg, mt, kt, ln); },
           sWp, sB, sTmp, nullptr, 64, tid);
  __syncthreads();
  dense256<32, 32, true>([&](int mt, int kt, int ln) -> v16h {
             return loadAfrag<32, 64>(sTmp, mt, kt, ln); },
           sWp + 1024, sB + 32, sTmp + 32, nullptr, 64, tid);
  __syncthreads();
  dense256<32, 64, false>([&](int mt, int kt, int ln) -> v16h {
             return loadAfrag<32, 64>(sTmp + 32, mt, kt, ln); },
           sWp + 2048, sB + 64, nullptr, sAcc, 64, tid);
  inormStats(sAcc, PP, 64, 64, sRed, tid);
  for (int i = tid; i < PP * 64; i += 256) {
    int c = i & 63;
    sXb[i] = (h16)((sAcc[i] - sRed[c]) * sRed[64 + c]);   // -> x1
  }
  __syncthreads();

  // ---------- two GINE convs ----------
  for (int pass = 0; pass < 2; ++pass) {
    const h16* xin = sXb + pass * XSTRIDE;
    h16* xout = sXb + (pass + 1) * XSTRIDE;
    const int wl = pass ? iL2W  : iL1W;
    const int wn = pass ? iG2W0 : iG1W0;
    packB<8, 64>(sWp,         a.W[wl],     tid);   // lin  -> 2048 halves @0
    packB<64, 32>(sWp + 2048, a.W[wn],     tid);   // nn1  -> 2048 @2048
    packB<32, 32>(sWp + 4096, a.W[wn + 2], tid);   // nn2  -> 1024 @4096
    packB<32, 64>(sWp + 5120, a.W[wn + 4], tid);   // nn3  -> 2048 @5120
    copyW(sB,       a.W[wl + 1], 64, tid);
    copyW(sB + 64,  a.W[wn + 1], 32, tid);
    copyW(sB + 96,  a.W[wn + 3], 32, tid);
    copyW(sB + 128, a.W[wn + 5], 64, tid);
    __syncthreads();
    for (int ch = 0; ch < 8; ++ch) {               // 256-edge chunks -> 32 nodes each
      const h16* Ein = sE + ch * 256 * 8;
      dense256<8, 64, false>([&](int mt, int kt, int ln) -> v16h {
                 return loadAfrag<8, 8>(Ein, mt, kt, ln); },
               sWp, sB, sMsg, nullptr, 64, tid);
      __syncthreads();
      for (int i = tid; i < 32 * 64; i += 256) {   // relu(x[col]+lin(e)) deg-sum
        int ln2 = i >> 6, c = i & 63;
        int n = ch * 32 + ln2;
        float s = 0.f;
#pragma unroll
        for (int d = 0; d < DEG; ++d) {
          int le  = ln2 * DEG + d;
          int src = sCol[n * DEG + d];
          float m = (float)xin[src * 64 + c] + (float)sMsg[le * 64 + c];
          s += fmaxf(m, 0.f);
        }
        sAcc[n * 64 + c] = s + (float)xin[n * 64 + c];
      }
      __syncthreads();
    }
    for (int i = tid; i < PP * 64; i += 256) sMsg[i] = (h16)sAcc[i];  // f16 A-staging
    __syncthreads();
    dense256<64, 32, true>([&](int mt, int kt, int ln) -> v16h {
               return loadAfrag<64, 64>(sMsg, mt, kt, ln); },
             sWp + 2048, sB + 64, sTmp, nullptr, 64, tid);
    __syncthreads();
    dense256<32, 32, true>([&](int mt, int kt, int ln) -> v16h {
               return loadAfrag<32, 64>(sTmp, mt, kt, ln); },
             sWp + 4096, sB + 96, sTmp + 32, nullptr, 64, tid);
    __syncthreads();
    dense256<32, 64, false>([&](int mt, int kt, int ln) -> v16h {
               return loadAfrag<32, 64>(sTmp + 32, mt, kt, ln); },
             sWp + 5120, sB + 128, nullptr, sAcc, 64, tid);
    inormStats(sAcc, PP, 64, 64, sRed, tid);
    for (int i = tid; i < PP * 64; i += 256) {
      int c = i & 63;
      xout[i] = (h16)((sAcc[i] - sRed[c]) * sRed[64 + c]);
    }
    __syncthreads();
  }

  // ---------- gate MLP on h=[x1|x2|x3] and softmax attention pooling ----------
  packB<192, 32>(sWp, a.W[iGaW0], tid);           // 6144 halves
  copyW(sB,      a.W[iGaB0], 32, tid);
  copyW(sB + 32, a.W[iGaW1], 32, tid);   // second layer weight [32,1]
  copyW(sB + 64, a.W[iGaB1], 1,  tid);
  __syncthreads();
  dense256<192, 32, true>([&](int mt, int kt, int ln) -> v16h {
             v16h aa;
             int m  = mt * 16 + (ln & 15);
             int kb = kt * 32 + ((ln >> 4) << 3);
#pragma unroll
             for (int i = 0; i < 16; ++i) {
               int k = kb + (i & 7) + ((i >> 3) << 4);   // k < 192 always
               aa[i] = sXb[(k >> 6) * XSTRIDE + m * 64 + (k & 63)];
             }
             return aa;
           },
           sWp, sB, sTmp, nullptr, 64, tid);
  __syncthreads();
  {
    float s = sB[64];
    const h16* hr = sTmp + tid * 64;
    for (int k = 0; k < 32; ++k) s += (float)hr[k] * sB[32 + k];
    sGate[tid] = s;
  }
  __syncthreads();
  sRed[tid] = sGate[tid];
  __syncthreads();
  for (int off = 128; off > 0; off >>= 1) {
    if (tid < off) sRed[tid] = fmaxf(sRed[tid], sRed[tid + off]);
    __syncthreads();
  }
  float gmax = sRed[0];
  __syncthreads();
  float w = __expf(sGate[tid] - gmax);
  sGate[tid] = w;
  sRed[tid] = w;
  __syncthreads();
  for (int off = 128; off > 0; off >>= 1) {
    if (tid < off) sRed[tid] += sRed[tid + off];
    __syncthreads();
  }
  float denom = sRed[0];
  __syncthreads();
  if (tid < 192) {
    const h16* hb = sXb + (tid >> 6) * XSTRIDE;
    int cc = tid & 63;
    float p = 0.f;
    for (int n2 = 0; n2 < PP; ++n2) p += sGate[n2] * (float)hb[n2 * 64 + cc];
    a.pooled[(size_t)g * 192 + tid] = p / denom;
  }
}

// ================= small cross-graph kernels =================
__global__ void bn_pooled_stats(const float* pooled, float* bnM, float* bnR) {
  int c = threadIdx.x;
  if (c >= 192) return;
  float s = 0.f, q = 0.f;
  for (int g2 = 0; g2 < GG; ++g2) { float v = pooled[g2 * 192 + c]; s += v; q += v * v; }
  float m = s / (float)GG, var = q / (float)GG - m * m;
  bnM[c] = m; bnR[c] = rsqrtf(var + EPSN);
}

__global__ void final_hidden(const float* pooled, const float* bnM, const float* bnR,
                             const float* scales, const float* W1, const float* b1,
                             float* t) {
  int g2 = blockIdx.x, j = threadIdx.x;   // 32 threads
  float s = b1[j];
  for (int c = 0; c < 192; ++c)
    s += (pooled[g2 * 192 + c] - bnM[c]) * bnR[c] * W1[c * 32 + j];
  s += logf(scales[g2] + 1e-5f) * W1[192 * 32 + j];
  t[g2 * 32 + j] = s;
}

__global__ void bn_hidden_stats(const float* t, float* hM, float* hR) {
  int j = threadIdx.x;
  if (j >= 32) return;
  float s = 0.f, q = 0.f;
  for (int g2 = 0; g2 < GG; ++g2) { float v = t[g2 * 32 + j]; s += v; q += v * v; }
  float m = s / (float)GG, var = q / (float)GG - m * m;
  hM[j] = m; hR[j] = rsqrtf(var + EPSN);
}

__global__ void final_out(const float* t, const float* hM, const float* hR,
                          const float* W2, const float* b2, float* out) {
  int g2 = blockIdx.x, o = threadIdx.x;   // 8 threads
  float s = b2[o];
  for (int j = 0; j < 32; ++j) {
    float hv = fmaxf((t[g2 * 32 + j] - hM[j]) * hR[j], 0.f);
    s += hv * W2[j * 8 + o];
  }
  out[g2 * 8 + o] = s;
}

// ================= host launch =================
extern "C" void kernel_launch(void* const* d_in, const int* in_sizes, int n_in,
                              void* d_out, int out_size, void* d_ws, size_t ws_size,
                              hipStream_t stream) {
  (void)in_sizes; (void)n_in; (void)out_size; (void)ws_size;
  K1Args a;
  a.x   = (const float*)d_in[0];
  a.ea  = (const float*)d_in[1];
  const float* scales = (const float*)d_in[2];
  // d_in[3] = row (implicit: row[e]=e/8), d_in[4] = col, d_in[5] = batch (implicit)
  a.col = (const int*)d_in[4];
  for (int i = 0; i < NW_TOTAL; ++i) a.W[i] = (const float*)d_in[6 + i];
  const float* fW1 = (const float*)d_in[44];  // final [193,32]
  const float* fb1 = (const float*)d_in[45];
  const float* fW2 = (const float*)d_in[46];  // final [32,8]
  const float* fb2 = (const float*)d_in[47];

  float* ws     = (float*)d_ws;
  float* pooled = ws;                       // [G,192]
  float* t      = pooled + (size_t)GG * 192;
  float* bnM    = t + (size_t)GG * 32;
  float* bnR    = bnM + 192;
  float* hM     = bnR + 192;
  float* hR     = hM + 32;
  a.pooled = pooled;

  trajs_graph_kernel<<<dim3(GG), dim3(256), LDS_BYTES, stream>>>(a);
  bn_pooled_stats<<<dim3(1), dim3(192), 0, stream>>>(pooled, bnM, bnR);
  final_hidden<<<dim3(GG), dim3(32), 0, stream>>>(pooled, bnM, bnR, scales, fW1, fb1, t);
  bn_hidden_stats<<<dim3(1), dim3(32), 0, stream>>>(t, hM, hR);
  final_out<<<dim3(GG), dim3(8), 0, stream>>>(t, hM, hR, fW2, fb2, (float*)d_out);
}
